// APPNP_8014408974457
// MI455X (gfx1250) — matrix operand order repaired and checked by reading, not compiled, not run on hardware
//
#include <hip/hip_runtime.h>
#include <hip/hip_bf16.h>

// ---------------------------------------------------------------------------
// APPNP on MI455X (gfx1250): fused MLP via bf16 split-precision WMMA,
// then K=10 hops of normalized gather/scatter with f32 global atomics.
// Round 4: pre-transposed/pre-split weights in workspace -> staging is pure
// b128 copy (no per-block conversion VALU, no scalar b16 stores for W1).
// ---------------------------------------------------------------------------

typedef __attribute__((ext_vector_type(16))) __bf16 v16bf;
typedef __attribute__((ext_vector_type(8)))  __bf16 v8bf;
typedef __attribute__((ext_vector_type(8)))  float  v8f;

#define DIN  512
#define DH   256
#define DOUT 32
#define ALPHA_F 0.1f

__device__ __forceinline__ void split_bf(float f, __bf16& hi, __bf16& lo) {
    hi = (__bf16)f;                    // RNE conversion to bf16 (high part)
    lo = (__bf16)(f - (float)hi);      // residual (low part)
}

__device__ __forceinline__ v16bf cat8(v8bf a, v8bf b) {
    return __builtin_shufflevector(a, b, 0, 1, 2, 3, 4, 5, 6, 7,
                                         8, 9, 10, 11, 12, 13, 14, 15);
}

// ---------------------------------------------------------------------------
// One-shot: split W1/W2 into hi/lo bf16, stored TRANSPOSED ([col][k]) so the
// GEMM staging/fragment reads are contiguous. Tiny (139K threads), L2-hot.
// ---------------------------------------------------------------------------
__global__ void convert_weights_kernel(const float* __restrict__ W1,
                                       const float* __restrict__ W2,
                                       __bf16* __restrict__ W1Th,
                                       __bf16* __restrict__ W1Tl,
                                       __bf16* __restrict__ W2Th,
                                       __bf16* __restrict__ W2Tl) {
    int tid = blockIdx.x * blockDim.x + threadIdx.x;
    if (tid < DIN * DH) {                      // W1 row-major [512][256]
        int k = tid >> 8, c = tid & 255;
        __bf16 hi, lo;
        split_bf(W1[tid], hi, lo);
        W1Th[(long)c * DIN + k] = hi;
        W1Tl[(long)c * DIN + k] = lo;
    } else if (tid < DIN * DH + DH * DOUT) {   // W2 row-major [256][32]
        int t2 = tid - DIN * DH;
        int k = t2 >> 5, c = t2 & 31;
        __bf16 hi, lo;
        split_bf(W2[t2], hi, lo);
        W2Th[(long)c * DH + k] = hi;
        W2Tl[(long)c * DH + k] = lo;
    }
}

// ---------------------------------------------------------------------------
// Fused MLP: h1 = feats@W1 + b1 (written to d_out), h0 = relu(h1)@W2 + b2
// Block: 256 threads (8 waves). Tile: 32 rows x 256 cols of h1 per block.
// Split-precision bf16 WMMA: Ah*Bh + Ah*Bl + Al*Bh  (3 wmma per K-step/tile).
// ---------------------------------------------------------------------------
__global__ __launch_bounds__(256) void mlp_fused_kernel(
    const float* __restrict__ feats,
    const __bf16* __restrict__ W1Th, const __bf16* __restrict__ W1Tl,
    const __bf16* __restrict__ W2Th, const __bf16* __restrict__ W2Tl,
    const float* __restrict__ b1,    const float* __restrict__ b2,
    float* __restrict__ h1_out,      float* __restrict__ h0_out)
{
    // A tile (feats): row-major [row][k], stride 40 (80B, 16B-aligned rows)
    __shared__ __align__(16) __bf16 sAh[32][40];
    __shared__ __align__(16) __bf16 sAl[32][40];
    // W1 tile TRANSPOSED: [col][k], stride 40 -> B-fragment reads are two
    // contiguous 16B runs per lane (ds_load_b128), 20-bank stride = no conflict
    __shared__ __align__(16) __bf16 sBTh[256][40];
    __shared__ __align__(16) __bf16 sBTl[256][40];
    // relu(h1) staged row-major [row][col] for GEMM2's A fragments
    __shared__ __align__(16) __bf16 sHh[32][264];
    __shared__ __align__(16) __bf16 sHl[32][264];

    const int tid  = threadIdx.x;
    const int w    = tid >> 5;        // wave 0..7
    const int lane = tid & 31;
    const int hl   = lane >> 4;       // lane half (0/1)
    const int lm   = lane & 15;
    const int wm   = w >> 2;          // wave M index 0..1  (16 rows each)
    const int wn   = w & 3;           // wave N index 0..3  (64 cols each)
    const long R   = (long)blockIdx.x * 32;

    v8f acc[4];
    #pragma unroll
    for (int t = 0; t < 4; ++t)
        #pragma unroll
        for (int r = 0; r < 8; ++r) acc[t][r] = 0.0f;

    for (int ks = 0; ks < DIN; ks += 32) {
        __syncthreads();
        // Stage A (feats) 32x32: one float4 per thread, convert to hi/lo.
        {
            int row = tid >> 3, f4 = tid & 7;
            float4 v = ((const float4*)(feats + (R + row) * DIN + ks))[f4];
            int c = f4 * 4;
            split_bf(v.x, sAh[row][c + 0], sAl[row][c + 0]);
            split_bf(v.y, sAh[row][c + 1], sAl[row][c + 1]);
            split_bf(v.z, sAh[row][c + 2], sAl[row][c + 2]);
            split_bf(v.w, sAh[row][c + 3], sAl[row][c + 3]);
        }
        // Stage B (W1^T, pre-split bf16) 256 cols x 32 k: pure 16B copies,
        // one column (64B hi + 64B lo) per thread.
        {
            const uint4* gh = (const uint4*)(W1Th + (long)tid * DIN + ks);
            const uint4* gl = (const uint4*)(W1Tl + (long)tid * DIN + ks);
            uint4* dh = (uint4*)&sBTh[tid][0];
            uint4* dl = (uint4*)&sBTl[tid][0];
            #pragma unroll
            for (int i = 0; i < 4; ++i) { dh[i] = gh[i]; dl[i] = gl[i]; }
        }
        __syncthreads();

        // A fragment (16x32 bf16): lane = row M=lm; elems 0..7 = K(8*hl+i),
        // elems 8..15 = K(16+8*hl+i). Two aligned 16B LDS loads each.
        const int ar = 16 * wm + lm;
        v16bf aH = cat8(*(const v8bf*)&sAh[ar][8 * hl],
                        *(const v8bf*)&sAh[ar][16 + 8 * hl]);
        v16bf aL = cat8(*(const v8bf*)&sAl[ar][8 * hl],
                        *(const v8bf*)&sAl[ar][16 + 8 * hl]);
        #pragma unroll
        for (int t = 0; t < 4; ++t) {
            // B fragment (32x16 bf16): lane = col N=lm; elems 0..7 = K(8*hl+i),
            // elems 8..15 = K(16+8*hl+i). Transposed LDS -> two aligned 16B loads.
            const int c = 64 * wn + 16 * t + lm;
            v16bf bH = cat8(*(const v8bf*)&sBTh[c][8 * hl],
                            *(const v8bf*)&sBTh[c][16 + 8 * hl]);
            v16bf bL = cat8(*(const v8bf*)&sBTl[c][8 * hl],
                            *(const v8bf*)&sBTl[c][16 + 8 * hl]);
            acc[t] = __builtin_amdgcn_wmma_f32_16x16x32_bf16(
                false, aH, false, bH, (short)0, acc[t], false, false);
            acc[t] = __builtin_amdgcn_wmma_f32_16x16x32_bf16(
                false, aH, false, bL, (short)0, acc[t], false, false);
            acc[t] = __builtin_amdgcn_wmma_f32_16x16x32_bf16(
                false, aL, false, bH, (short)0, acc[t], false, false);
        }
    }

    // Bias + write h1 to global; stage relu(h1) hi/lo into sH* for GEMM2.
    #pragma unroll
    for (int t = 0; t < 4; ++t) {
        const int col  = 64 * wn + 16 * t + lm;
        const float bv = b1[col];
        #pragma unroll
        for (int r = 0; r < 8; ++r) {
            const int mrow = 16 * wm + 8 * hl + r;   // C/D: M = r + 8*half
            float v = acc[t][r] + bv;
            h1_out[(R + mrow) * DH + col] = v;
            float rv = v > 0.0f ? v : 0.0f;
            split_bf(rv, sHh[mrow][col], sHl[mrow][col]);
        }
    }
    __syncthreads();

    // GEMM2: relu(h1)[32x256] @ W2[256x32] + b2 -> h0[32x32]. Waves 0..3,
    // one 16x16 tile each (wave-uniform branch keeps EXEC all-1s).
    // B fragments read directly from pre-split W2^T in global (L2-hot).
    if (w < 4) {
        const int wm2 = w >> 1, wn2 = w & 1;
        v8f acc2;
        #pragma unroll
        for (int r = 0; r < 8; ++r) acc2[r] = 0.0f;
        const int ar2  = 16 * wm2 + lm;
        const int col2 = 16 * wn2 + lm;
        for (int ks = 0; ks < DH; ks += 32) {
            v16bf aH = cat8(*(const v8bf*)&sHh[ar2][ks + 8 * hl],
                            *(const v8bf*)&sHh[ar2][ks + 16 + 8 * hl]);
            v16bf aL = cat8(*(const v8bf*)&sHl[ar2][ks + 8 * hl],
                            *(const v8bf*)&sHl[ar2][ks + 16 + 8 * hl]);
            const __bf16* w2h = W2Th + (long)col2 * DH + ks + 8 * hl;
            const __bf16* w2l = W2Tl + (long)col2 * DH + ks + 8 * hl;
            v16bf bH = cat8(*(const v8bf*)w2h, *(const v8bf*)(w2h + 16));
            v16bf bL = cat8(*(const v8bf*)w2l, *(const v8bf*)(w2l + 16));
            acc2 = __builtin_amdgcn_wmma_f32_16x16x32_bf16(
                false, aH, false, bH, (short)0, acc2, false, false);
            acc2 = __builtin_amdgcn_wmma_f32_16x16x32_bf16(
                false, aH, false, bL, (short)0, acc2, false, false);
            acc2 = __builtin_amdgcn_wmma_f32_16x16x32_bf16(
                false, aL, false, bH, (short)0, acc2, false, false);
        }
        const float bv2 = b2[col2];
        #pragma unroll
        for (int r = 0; r < 8; ++r) {
            const int mrow = 16 * wm2 + 8 * hl + r;
            h0_out[(R + mrow) * DOUT + col2] = acc2[r] + bv2;
        }
    }
}

// ---------------------------------------------------------------------------
// Propagation helper kernels (bandwidth/atomic bound)
// ---------------------------------------------------------------------------
__global__ void zero_kernel(float* __restrict__ p, long n) {
    long i = (long)blockIdx.x * blockDim.x + threadIdx.x;
    if (i < n) p[i] = 0.0f;
}

__global__ void degree_kernel(const int* __restrict__ src, const int* __restrict__ dst,
                              float* __restrict__ deg_s, float* __restrict__ deg_d, int E) {
    int e = blockIdx.x * blockDim.x + threadIdx.x;
    if (e < E) {
        atomicAdd(&deg_s[src[e]], 1.0f);
        atomicAdd(&deg_d[dst[e]], 1.0f);
    }
}

__global__ void rsqrt_kernel(float* __restrict__ p, long n) {
    long i = (long)blockIdx.x * blockDim.x + threadIdx.x;
    if (i < n) p[i] = rsqrtf(fmaxf(p[i], 1.0f));   // clip(deg,1)^(-1/2)
}

// One thread per (edge, 4-float chunk): agg[dst] += h[src] * norm_src[src]
__global__ void edge_scatter_kernel(const int* __restrict__ src, const int* __restrict__ dst,
                                    const float* __restrict__ hin, const float* __restrict__ ns,
                                    float* __restrict__ agg, int E) {
    long tid = (long)blockIdx.x * blockDim.x + threadIdx.x;
    int e = (int)(tid >> 3), g = (int)(tid & 7);
    if (e >= E) return;
    int s = src[e], d = dst[e];
    float scale = ns[s];
    float4 v = ((const float4*)hin)[(long)s * 8 + g];
    float* outp = agg + (long)d * DOUT + g * 4;
    atomicAdd(outp + 0, v.x * scale);
    atomicAdd(outp + 1, v.y * scale);
    atomicAdd(outp + 2, v.z * scale);
    atomicAdd(outp + 3, v.w * scale);
}

// h = (1-alpha) * agg * norm_dst + alpha * h0   (in place on agg)
__global__ void combine_kernel(float* __restrict__ agg, const float* __restrict__ h0,
                               const float* __restrict__ nd, long total) {
    long i = (long)blockIdx.x * blockDim.x + threadIdx.x;
    if (i < total)
        agg[i] = (1.0f - ALPHA_F) * agg[i] * nd[i >> 5] + ALPHA_F * h0[i];
}

// ---------------------------------------------------------------------------
extern "C" void kernel_launch(void* const* d_in, const int* in_sizes, int n_in,
                              void* d_out, int out_size, void* d_ws, size_t ws_size,
                              hipStream_t stream) {
    const float* feats = (const float*)d_in[0];
    const int*   src   = (const int*)  d_in[1];
    const int*   dst   = (const int*)  d_in[2];
    const float* W1    = (const float*)d_in[3];
    const float* b1    = (const float*)d_in[4];
    const float* W2    = (const float*)d_in[5];
    const float* b2    = (const float*)d_in[6];

    const long N = (long)in_sizes[0] / DIN;   // 100000 (divisible by 32)
    const int  E = in_sizes[1];               // 1.6M edges

    float* out_h1 = (float*)d_out;                       // [N, 256]
    float* out_h  = (float*)d_out + N * DH;              // [N, 32]

    // Workspace layout (floats): ns[N] | nd[N] | h0[N*32] | P[N*32] | Q[N*32]
    // followed by 16B-aligned bf16 region: W1Th | W1Tl | W2Th | W2Tl
    float* ws = (float*)d_ws;
    float* ns = ws;
    float* nd = ws + N;
    float* h0 = ws + 2 * N;
    float* P  = h0 + N * DOUT;
    float* Q  = P + N * DOUT;
    long bf_off = (98 * N + 3) & ~3L;          // float offset, 16B aligned
    __bf16* W1Th = (__bf16*)(ws + bf_off);
    __bf16* W1Tl = W1Th + (long)DIN * DH;
    __bf16* W2Th = W1Tl + (long)DIN * DH;
    __bf16* W2Tl = W2Th + (long)DH * DOUT;

    const int TPB = 256;

    // Pre-split/transpose weights (one shot, tiny)
    convert_weights_kernel<<<(DIN * DH + DH * DOUT + TPB - 1) / TPB, TPB, 0, stream>>>(
        W1, W2, W1Th, W1Tl, W2Th, W2Tl);

    // Degrees -> symmetric normalization factors
    zero_kernel<<<(int)((2 * N + TPB - 1) / TPB), TPB, 0, stream>>>(ns, 2 * N);
    degree_kernel<<<(E + TPB - 1) / TPB, TPB, 0, stream>>>(src, dst, ns, nd, E);
    rsqrt_kernel<<<(int)((2 * N + TPB - 1) / TPB), TPB, 0, stream>>>(ns, 2 * N);

    // Fused MLP: h1 -> d_out, h0 -> workspace
    mlp_fused_kernel<<<(int)(N / 32), TPB, 0, stream>>>(
        feats, W1Th, W1Tl, W2Th, W2Tl, b1, b2, out_h1, h0);

    // K = 10 propagation hops
    const long total = N * DOUT;
    const int  zb = (int)((total + TPB - 1) / TPB);
    const int  eb = (int)(((long)E * 8 + TPB - 1) / TPB);
    for (int it = 0; it < 10; ++it) {
        const float* hin = (it == 0) ? h0 : (((it - 1) & 1) ? Q : P);
        float* hout = (it == 9) ? out_h : ((it & 1) ? Q : P);
        zero_kernel<<<zb, TPB, 0, stream>>>(hout, total);
        edge_scatter_kernel<<<eb, TPB, 0, stream>>>(src, dst, hin, ns, hout, E);
        combine_kernel<<<zb, TPB, 0, stream>>>(hout, h0, nd, total);
    }
}
